// Window_Attention_Intent_Decoder_22797686408103
// MI455X (gfx1250) — matrix operand order repaired and checked by reading, not compiled, not run on hardware
//
#include <hip/hip_runtime.h>
#include <hip/hip_bf16.h>

// ---------------- constants (from reference) ----------------
// B=16, S=512, D=768, W=3, NH=4, HD=192, DFF=2048, NI=18
// N = S-W+1 = 510 ; M = B*N = 8160 ; R = M*W = 24480
#define cB   16
#define cS   512
#define cD   768
#define cW   3
#define cNH  4
#define cHD  192
#define cDFF 2048
#define cNI  18
#define cN   510
#define cM   8160
#define cR   24480

#define OFF_WNUM  (cB * cN * cNI)          // 146880
#define OFF_ALIGN (OFF_WNUM + cB)          // 146896

// ---------------- WMMA types / helpers ----------------
typedef __attribute__((ext_vector_type(16))) __bf16 v16bf;
typedef __attribute__((ext_vector_type(8)))  float  v8f;

union Frag { v16bf v; uint4 u[2]; };

__device__ inline unsigned short f2bf(float f) {
  unsigned u = __float_as_uint(f);
  unsigned r = u + 0x7fffu + ((u >> 16) & 1u);   // round-to-nearest-even
  return (unsigned short)(r >> 16);
}
__device__ inline float bf2f(unsigned short s) {
  return __uint_as_float(((unsigned)s) << 16);
}

// A fragment (16x32 bf16, row-major source [16, K]): lane r=lane&15 is row M,
// half h=lane>>4: elems 0..7 = K = 8h..8h+7 ; elems 8..15 = K = 16+8h..+7
__device__ inline v16bf load_frag_a(const unsigned short* __restrict__ p, int hl) {
  Frag f;
  const unsigned short* q = p + 8 * hl;
  f.u[0] = *(const uint4*)(q);
  f.u[1] = *(const uint4*)(q + 16);
  return f.v;
}
// B fragment (32x16 bf16 from row-major W[N,K]): lane r = column N,
// half h: elems 0..7 = K = 8h..8h+7 ; elems 8..15 = K = 16+8h..+7
__device__ inline v16bf load_frag_b(const unsigned short* __restrict__ p, int hl) {
  Frag f;
  const unsigned short* q = p + 8 * hl;
  f.u[0] = *(const uint4*)(q);
  f.u[1] = *(const uint4*)(q + 16);
  return f.v;
}

__device__ inline v8f wmma_bf16(v16bf a, v16bf b, v8f c) {
  return __builtin_amdgcn_wmma_f32_16x16x32_bf16(false, a, false, b, (short)0, c,
                                                 false, false);
}

// ---------------- cast kernels ----------------
__global__ void cast_f32_bf16(const float* __restrict__ src,
                              unsigned short* __restrict__ dst, int n) {
  int i = blockIdx.x * blockDim.x + threadIdx.x;
  if (i < n) dst[i] = f2bf(src[i]);
}

__global__ void pad_intent_w(const float* __restrict__ src,
                             unsigned short* __restrict__ dst) {  // 32x768
  int i = blockIdx.x * blockDim.x + threadIdx.x;
  if (i < 32 * cD) {
    int r = i / cD;
    dst[i] = (r < cNI) ? f2bf(src[i]) : (unsigned short)0;
  }
}

// ---------------- main WMMA GEMM: C = A @ Bw^T (+ epilogue) ----------------
// A [Mdim,K] bf16 row-major, Bw [Ndim,K] bf16 row-major. Ndim multiple of 256.
// Block = 256 threads = 8 waves (2 along M, 4 along N) -> 128x256 C tile.
// Each wave owns 64x64 (4x4 fragments): per K=32 step, 8 fragment loads feed
// 16 WMMAs (load:wmma = 0.5), 4x register reuse of A and B operands.
#define EPI_BF16   0   // out bf16 = acc + bias
#define EPI_RESHID 1   // out f32  = acc + bias + hidden[b, n+t, col]   (row = (m,t))
#define EPI_RELU   2   // out bf16 = relu(acc + bias)
#define EPI_RESBF  3   // out f32  = acc + bias + bf16-resid[row, col]

template <int EPI>
__global__ __launch_bounds__(256) void gemm_bf16_wmma(
    const unsigned short* __restrict__ A, const unsigned short* __restrict__ Bw,
    const float* __restrict__ bias, const float* __restrict__ residF,
    const unsigned short* __restrict__ residB, float* __restrict__ outF,
    unsigned short* __restrict__ outB, int Mdim, int Ndim, int K) {
  int wid = threadIdx.x >> 5, lane = threadIdx.x & 31;
  int wm = wid & 1, wn = wid >> 1;                 // 2 waves on M, 4 on N
  int m0 = blockIdx.y * 128 + wm * 64;
  int n0 = blockIdx.x * 256 + wn * 64;
  int rl = lane & 15, hl = lane >> 4;

  v8f acc[4][4] = {};
  int rowA[4];
#pragma unroll
  for (int i = 0; i < 4; i++) {
    int row = m0 + i * 16 + rl;
    rowA[i] = (row < Mdim) ? row : (Mdim - 1);   // clamp; store is guarded
  }
  const unsigned short* colB[4];
#pragma unroll
  for (int j = 0; j < 4; j++) colB[j] = Bw + (size_t)(n0 + j * 16 + rl) * K;

  for (int k0 = 0; k0 < K; k0 += 32) {
    v16bf a[4], b[4];
#pragma unroll
    for (int i = 0; i < 4; i++)
      a[i] = load_frag_a(A + (size_t)rowA[i] * K + k0, hl);
#pragma unroll
    for (int j = 0; j < 4; j++) b[j] = load_frag_b(colB[j] + k0, hl);
    if (k0 + 32 < K) {
      __builtin_prefetch(A + (size_t)rowA[0] * K + k0 + 32, 0, 0);
      __builtin_prefetch(colB[0] + k0 + 32, 0, 0);
    }
#pragma unroll
    for (int i = 0; i < 4; i++)
#pragma unroll
      for (int j = 0; j < 4; j++) acc[i][j] = wmma_bf16(a[i], b[j], acc[i][j]);
  }

  // epilogue: C element (VGPR e, lane) -> row = m0+i*16+e+8*hl, col = n0+j*16+rl
#pragma unroll
  for (int i = 0; i < 4; i++) {
#pragma unroll
    for (int j = 0; j < 4; j++) {
      int col = n0 + j * 16 + rl;
      float bv = bias ? bias[col] : 0.f;
#pragma unroll
      for (int e = 0; e < 8; e++) {
        int row = m0 + i * 16 + e + 8 * hl;
        if (row < Mdim) {
          float v = acc[i][j][e] + bv;
          if constexpr (EPI == EPI_RESHID) {
            int mw = row / 3, t = row - 3 * mw;
            int bb = mw / cN, nn = mw - bb * cN;
            v += residF[((size_t)(bb * cS + nn + t)) * cD + col];
          } else if constexpr (EPI == EPI_RESBF) {
            v += bf2f(residB[(size_t)row * Ndim + col]);
          } else if constexpr (EPI == EPI_RELU) {
            v = v > 0.f ? v : 0.f;
          }
          if constexpr (EPI == EPI_BF16 || EPI == EPI_RELU)
            outB[(size_t)row * Ndim + col] = f2bf(v);
          else
            outF[(size_t)row * Ndim + col] = v;
        }
      }
    }
  }
}

// ---------------- attention: one wave per window ----------------
// qkv rows are per token: [q(768) | k(768) | v(768)], head h owns dims h*192..+191.
__global__ __launch_bounds__(256) void attn_win_kernel(
    const unsigned short* __restrict__ qkv, unsigned short* __restrict__ o) {
  int wid = threadIdx.x >> 5, lane = threadIdx.x & 31;
  int m = blockIdx.x * 8 + wid;
  if (m >= cM) return;
  int b = m / cN, n = m - b * cN;
  size_t base0 = ((size_t)(b * cS + n)) * (3 * cD);
  const float scale = 0.07216878364870323f;  // 1/sqrt(192)

  for (int h = 0; h < cNH; h++) {
    int off = h * cHD + lane * 6;  // 192/32 = 6 dims per lane
    float qv[3][6], kv[3][6], vv[3][6];
#pragma unroll
    for (int t = 0; t < 3; t++) {
      const unsigned* pq = (const unsigned*)(qkv + base0 + (size_t)t * (3 * cD) + off);
      const unsigned* pk = (const unsigned*)(qkv + base0 + (size_t)t * (3 * cD) + cD + off);
      const unsigned* pv = (const unsigned*)(qkv + base0 + (size_t)t * (3 * cD) + 2 * cD + off);
#pragma unroll
      for (int w = 0; w < 3; w++) {
        unsigned uq = pq[w], uk = pk[w], uv = pv[w];
        qv[t][2 * w] = bf2f((unsigned short)uq);
        qv[t][2 * w + 1] = bf2f((unsigned short)(uq >> 16));
        kv[t][2 * w] = bf2f((unsigned short)uk);
        kv[t][2 * w + 1] = bf2f((unsigned short)(uk >> 16));
        vv[t][2 * w] = bf2f((unsigned short)uv);
        vv[t][2 * w + 1] = bf2f((unsigned short)(uv >> 16));
      }
    }
    float sc[3][3];
#pragma unroll
    for (int i = 0; i < 3; i++)
#pragma unroll
      for (int j = 0; j < 3; j++) {
        float p = 0.f;
#pragma unroll
        for (int e = 0; e < 6; e++) p += qv[i][e] * kv[j][e];
#pragma unroll
        for (int d = 16; d >= 1; d >>= 1) p += __shfl_xor(p, d, 32);
        sc[i][j] = p * scale;
      }
#pragma unroll
    for (int i = 0; i < 3; i++) {
      float mx = fmaxf(sc[i][0], fmaxf(sc[i][1], sc[i][2]));
      float e0 = __expf(sc[i][0] - mx), e1 = __expf(sc[i][1] - mx),
            e2 = __expf(sc[i][2] - mx);
      float inv = 1.f / (e0 + e1 + e2);
      float a0 = e0 * inv, a1 = e1 * inv, a2 = e2 * inv;
      unsigned short* po = o + ((size_t)(m * 3 + i)) * cD + h * cHD + lane * 6;
#pragma unroll
      for (int e = 0; e < 6; e++)
        po[e] = f2bf(a0 * vv[0][e] + a1 * vv[1][e] + a2 * vv[2][e]);
    }
  }
}

// ---------------- LayerNorm helpers / kernels ----------------
__device__ inline float block_sum(float v, float* sh) {
#pragma unroll
  for (int d = 16; d >= 1; d >>= 1) v += __shfl_xor(v, d, 32);
  int wid = threadIdx.x >> 5;
  __syncthreads();
  if ((threadIdx.x & 31) == 0) sh[wid] = v;
  __syncthreads();
  float s = 0.f;
#pragma unroll
  for (int i = 0; i < 8; i++) s += sh[i];
  return s;
}

__global__ __launch_bounds__(256) void ln_bf16_kernel(
    const float* __restrict__ in, const float* __restrict__ g,
    const float* __restrict__ bta, unsigned short* __restrict__ outB, int rows) {
  __shared__ float sh[8];
  int row = blockIdx.x;
  if (row >= rows) return;
  const float* x = in + (size_t)row * cD;
  float lx[3], s = 0.f;
#pragma unroll
  for (int c = 0; c < 3; c++) {
    lx[c] = x[threadIdx.x + 256 * c];
    s += lx[c];
  }
  float mean = block_sum(s, sh) * (1.f / cD);
  float q = 0.f;
#pragma unroll
  for (int c = 0; c < 3; c++) {
    float d = lx[c] - mean;
    q += d * d;
  }
  float var = block_sum(q, sh) * (1.f / cD);
  float rs = rsqrtf(var + 1e-5f);
#pragma unroll
  for (int c = 0; c < 3; c++) {
    int col = threadIdx.x + 256 * c;
    outB[(size_t)row * cD + col] = f2bf((lx[c] - mean) * rs * g[col] + bta[col]);
  }
}

// LN2 + max-pool over W=3 + direct align writes (+ bf16 pooled for intent GEMM)
__global__ __launch_bounds__(256) void ln2_pool_kernel(
    const float* __restrict__ in, const float* __restrict__ g,
    const float* __restrict__ bta, unsigned short* __restrict__ pooledB,
    float* __restrict__ alignOut) {
  __shared__ float sh[8];
  int m = blockIdx.x;
  float mx[3] = {-1e30f, -1e30f, -1e30f};
  for (int t = 0; t < cW; t++) {
    const float* x = in + (size_t)(m * cW + t) * cD;
    float lx[3], s = 0.f;
#pragma unroll
    for (int c = 0; c < 3; c++) {
      lx[c] = x[threadIdx.x + 256 * c];
      s += lx[c];
    }
    float mean = block_sum(s, sh) * (1.f / cD);
    float q = 0.f;
#pragma unroll
    for (int c = 0; c < 3; c++) {
      float d = lx[c] - mean;
      q += d * d;
    }
    float var = block_sum(q, sh) * (1.f / cD);
    float rs = rsqrtf(var + 1e-5f);
#pragma unroll
    for (int c = 0; c < 3; c++) {
      int col = threadIdx.x + 256 * c;
      float y = (lx[c] - mean) * rs * g[col] + bta[col];
      mx[c] = fmaxf(mx[c], y);
    }
  }
  int b = m / cN, n = m - b * cN;
#pragma unroll
  for (int c = 0; c < 3; c++) {
    int col = threadIdx.x + 256 * c;
    pooledB[(size_t)m * cD + col] = f2bf(mx[c]);
    alignOut[((size_t)(b * (cN + 2)) + (n + 1)) * cD + col] = mx[c];
    if (n == 0) alignOut[((size_t)(b * (cN + 2))) * cD + col] = mx[c];
    if (n == cN - 1) alignOut[((size_t)(b * (cN + 2)) + cN + 1) * cD + col] = mx[c];
  }
}

// ---------------- intent head: [8160,768] x [32(18),768]^T ----------------
__global__ __launch_bounds__(256) void gemm_intent_wmma(
    const unsigned short* __restrict__ A, const unsigned short* __restrict__ Bw,
    const float* __restrict__ bias, float* __restrict__ pred) {
  int wid = threadIdx.x >> 5, lane = threadIdx.x & 31;
  int rl = lane & 15, hl = lane >> 4;
  int m0 = (blockIdx.x * 8 + wid) * 16;
  v8f acc[2] = {};
  int row = m0 + rl;
  if (row >= cM) row = cM - 1;
  for (int k0 = 0; k0 < cD; k0 += 32) {
    v16bf a = load_frag_a(A + (size_t)row * cD + k0, hl);
#pragma unroll
    for (int j = 0; j < 2; j++) {
      v16bf b = load_frag_b(Bw + (size_t)(j * 16 + rl) * cD + k0, hl);
      acc[j] = wmma_bf16(a, b, acc[j]);
    }
  }
#pragma unroll
  for (int j = 0; j < 2; j++) {
    int col = j * 16 + rl;
#pragma unroll
    for (int e = 0; e < 8; e++) {
      int r = m0 + e + 8 * hl;
      if (r < cM && col < cNI) pred[(size_t)r * cNI + col] = acc[j][e] + bias[col];
    }
  }
}

// ---------------- window_num ----------------
__global__ void wnum_kernel(const int* __restrict__ seq_lens, float* __restrict__ out) {
  int b = threadIdx.x;
  if (b < cB) {
    int wn = seq_lens[b] - (cW - 1);
    if (wn <= 0) wn = 1;
    out[b] = (float)wn;
  }
}

// ---------------- launcher ----------------
extern "C" void kernel_launch(void* const* d_in, const int* in_sizes, int n_in,
                              void* d_out, int out_size, void* d_ws, size_t ws_size,
                              hipStream_t stream) {
  (void)in_sizes; (void)n_in; (void)out_size; (void)ws_size;
  const float* hidden     = (const float*)d_in[0];
  const int*   seq_lens   = (const int*)d_in[1];
  const float* in_proj_w  = (const float*)d_in[2];
  const float* in_proj_b  = (const float*)d_in[3];
  const float* out_proj_w = (const float*)d_in[4];
  const float* out_proj_b = (const float*)d_in[5];
  const float* ln1_g      = (const float*)d_in[6];
  const float* ln1_b      = (const float*)d_in[7];
  const float* ln2_g      = (const float*)d_in[8];
  const float* ln2_b      = (const float*)d_in[9];
  const float* ffn_w1     = (const float*)d_in[10];
  const float* ffn_b1     = (const float*)d_in[11];
  const float* ffn_w2     = (const float*)d_in[12];
  const float* ffn_b2     = (const float*)d_in[13];
  const float* intent_w   = (const float*)d_in[14];
  const float* intent_b   = (const float*)d_in[15];
  float* out = (float*)d_out;

  // bump allocator over workspace (assumes ws_size >= ~325 MB)
  char* ws = (char*)d_ws;
  size_t cur = 0;
  auto alloc = [&](size_t bytes) -> char* {
    char* p = ws + cur;
    cur = (cur + bytes + 255) & ~(size_t)255;
    return p;
  };
  const int TOK = cB * cS;  // 8192 tokens
  unsigned short* hid_b  = (unsigned short*)alloc((size_t)TOK * cD * 2);
  unsigned short* win_b  = (unsigned short*)alloc((size_t)3 * cD * cD * 2);
  unsigned short* wout_b = (unsigned short*)alloc((size_t)cD * cD * 2);
  unsigned short* w1_b   = (unsigned short*)alloc((size_t)cDFF * cD * 2);
  unsigned short* w2_b   = (unsigned short*)alloc((size_t)cD * cDFF * 2);
  unsigned short* wi_b   = (unsigned short*)alloc((size_t)32 * cD * 2);
  unsigned short* qkv_b  = (unsigned short*)alloc((size_t)TOK * 3 * cD * 2);
  unsigned short* o_b    = (unsigned short*)alloc((size_t)cR * cD * 2);
  unsigned short* x1_b   = (unsigned short*)alloc((size_t)cR * cD * 2);
  unsigned short* h_b    = (unsigned short*)alloc((size_t)cR * cDFF * 2);
  float*          tmp_f  = (float*)alloc((size_t)cR * cD * 4);   // reused pre-LN1 & pre-LN2
  unsigned short* pool_b = (unsigned short*)alloc((size_t)cM * cD * 2);

  auto cdiv = [](int a, int b) { return (a + b - 1) / b; };

  // 1) f32 -> bf16 casts
  cast_f32_bf16<<<cdiv(TOK * cD, 256), 256, 0, stream>>>(hidden, hid_b, TOK * cD);
  cast_f32_bf16<<<cdiv(3 * cD * cD, 256), 256, 0, stream>>>(in_proj_w, win_b, 3 * cD * cD);
  cast_f32_bf16<<<cdiv(cD * cD, 256), 256, 0, stream>>>(out_proj_w, wout_b, cD * cD);
  cast_f32_bf16<<<cdiv(cDFF * cD, 256), 256, 0, stream>>>(ffn_w1, w1_b, cDFF * cD);
  cast_f32_bf16<<<cdiv(cD * cDFF, 256), 256, 0, stream>>>(ffn_w2, w2_b, cD * cDFF);
  pad_intent_w<<<cdiv(32 * cD, 256), 256, 0, stream>>>(intent_w, wi_b);

  // 2) QKV once per token (dedup across overlapping windows): [8192,768]x[768,2304]
  gemm_bf16_wmma<EPI_BF16><<<dim3((3 * cD) / 256, cdiv(TOK, 128)), 256, 0, stream>>>(
      hid_b, win_b, in_proj_b, nullptr, nullptr, nullptr, qkv_b, TOK, 3 * cD, cD);

  // 3) per-window attention -> o rows [24480,768] bf16
  attn_win_kernel<<<cM / 8, 256, 0, stream>>>(qkv_b, o_b);

  // 4) out-proj + hidden residual: [24480,768]x[768,768] -> tmp_f (f32)
  gemm_bf16_wmma<EPI_RESHID><<<dim3(cD / 256, cdiv(cR, 128)), 256, 0, stream>>>(
      o_b, wout_b, out_proj_b, hidden, nullptr, tmp_f, nullptr, cR, cD, cD);

  // 5) LN1 -> x1 bf16
  ln_bf16_kernel<<<cR, 256, 0, stream>>>(tmp_f, ln1_g, ln1_b, x1_b, cR);

  // 6) FFN1 + ReLU: [24480,768]x[768,2048] -> h bf16
  gemm_bf16_wmma<EPI_RELU><<<dim3(cDFF / 256, cdiv(cR, 128)), 256, 0, stream>>>(
      x1_b, w1_b, ffn_b1, nullptr, nullptr, nullptr, h_b, cR, cDFF, cD);

  // 7) FFN2 + x1 residual: [24480,2048]x[2048,768] -> tmp_f (f32, reuse)
  gemm_bf16_wmma<EPI_RESBF><<<dim3(cD / 256, cdiv(cR, 128)), 256, 0, stream>>>(
      h_b, w2_b, ffn_b2, nullptr, x1_b, tmp_f, nullptr, cR, cD, cDFF);

  // 8) LN2 + W=3 max-pool + align output (with first/last duplication)
  ln2_pool_kernel<<<cM, 256, 0, stream>>>(tmp_f, ln2_g, ln2_b, pool_b,
                                          out + OFF_ALIGN);

  // 9) intent head: [8160,768]x[768,32(18 valid)] -> pred_intent
  gemm_intent_wmma<<<cdiv(cM, 128), 256, 0, stream>>>(pool_b, wi_b, intent_b, out);

  // 10) window_num (as float values)
  wnum_kernel<<<1, 32, 0, stream>>>(seq_lens, out + OFF_WNUM);
}
